// multi_att_47132971106440
// MI455X (gfx1250) — compile-verified, matrix operand order checked by reading, not run on hardware
//
#include <hip/hip_runtime.h>
#include <hip/hip_bf16.h>

// Problem constants (shapes fixed by setup_inputs)
#define Bb   2
#define Cc   256
#define HWn  6400
#define CIM  32
#define PP   9

typedef __attribute__((ext_vector_type(16))) __bf16 v16bf;
typedef __attribute__((ext_vector_type(8)))  float  v8f;

union FragAB { v16bf v; uint4 q[2]; unsigned short s[16]; };
union FragC  { v8f   v; float f[8]; };

__device__ __forceinline__ unsigned short f2bf(float f) {
    union { __bf16 b; unsigned short s; } u; u.b = (__bf16)f; return u.s;
}
__device__ __forceinline__ unsigned int pack2(unsigned short lo, unsigned short hi) {
    return (unsigned int)lo | ((unsigned int)hi << 16);
}

// branchless sorted-insert into ascending top-9 (tv[0] = current min)
__device__ __forceinline__ void topk_insert(float (&tv)[9], int (&ti)[9], float x, int ci0) {
    if (x > tv[0]) {
        float cv = x; int ci = ci0;
#pragma unroll
        for (int s = 0; s < 8; ++s) {
            float nx = tv[s + 1]; int ni = ti[s + 1];
            bool up = cv > nx;
            tv[s] = up ? nx : cv;
            ti[s] = up ? ni : ci;
            cv = up ? cv : nx;
            ci = up ? ci : ni;
        }
        tv[8] = cv; ti[8] = ci;
    }
}

// CDNA5 async global->LDS copy (ASYNCcnt-tracked, no VGPR data path).
// lds_off: wave-relative LDS byte address (low 32 bits of a __shared__ pointer).
__device__ __forceinline__ void async_load_b128(unsigned lds_off, const void* gaddr) {
    asm volatile("global_load_async_to_lds_b128 %0, %1, off"
                 :: "v"(lds_off), "v"((unsigned long long)(uintptr_t)gaddr)
                 : "memory");
}
__device__ __forceinline__ void wait_asynccnt0() {
    asm volatile("s_wait_asynccnt 0x0" ::: "memory");
}

// ---------------------------------------------------------------------------
// Kernel 1: fold weights.
//   W_eff[o][d] = sum_c mlp_w[o, c] * catt[c] * back_w[c, d]   (256x32, bf16)
//   W2  [o][c] = mlp_w[o, 256+c] * catt1[c]                    (256x256, bf16)
//   scaleo[o]  = gamma[o]/sqrt(1+eps); biaso[o] = mlp_b[o]*scaleo[o] + beta[o]
// ---------------------------------------------------------------------------
__global__ __launch_bounds__(256)
void precompute_kernel(const float* __restrict__ mlp_w, const float* __restrict__ catt,
                       const float* __restrict__ catt1, const float* __restrict__ back_w,
                       const float* __restrict__ mlp_b, const float* __restrict__ mlp_g,
                       const float* __restrict__ mlp_be,
                       unsigned short* __restrict__ Weff, unsigned short* __restrict__ W2,
                       float* __restrict__ scaleo, float* __restrict__ biaso)
{
    const int o = blockIdx.x;
    const int t = threadIdx.x;
    W2[o * 256 + t] = f2bf(mlp_w[o * 512 + 256 + t] * catt1[t]);
    if (t < CIM) {
        float s = 0.f;
        for (int c = 0; c < 256; ++c)
            s += mlp_w[o * 512 + c] * catt[c] * back_w[c * CIM + t];
        Weff[o * CIM + t] = f2bf(s);
    }
    if (t == 0) {
        float g = mlp_g[o] * rsqrtf(1.0f + 1e-5f);
        scaleo[o] = g;
        biaso[o]  = mlp_b[o] * g + mlp_be[o];
    }
}

// ---------------------------------------------------------------------------
// Kernel 2: q/k/v projections. One thread per pixel (coalesced along HW).
// Emits l2-normalized q,k as bf16 rows [B][HW][32]; v as f32 [B][HW][32].
// ---------------------------------------------------------------------------
__global__ __launch_bounds__(256)
void qkv_kernel(const float* __restrict__ feat,
                const float* __restrict__ Wq, const float* __restrict__ Wk,
                const float* __restrict__ Wv,
                unsigned short* __restrict__ qn, unsigned short* __restrict__ kn,
                float* __restrict__ vout)
{
    __shared__ float sW[3][128][32];          // 48 KB, weights staged in halves of C
    const int tid = threadIdx.x;
    const int g = blockIdx.x * 256 + tid;     // grid = 50 blocks -> exactly B*HW threads
    const int b = g / HWn, n = g % HWn;

    float4 aq4[8], ak4[8], av4[8];
#pragma unroll
    for (int i = 0; i < 8; ++i) {
        aq4[i] = make_float4(0.f, 0.f, 0.f, 0.f);
        ak4[i] = make_float4(0.f, 0.f, 0.f, 0.f);
        av4[i] = make_float4(0.f, 0.f, 0.f, 0.f);
    }
    const float* fbase = feat + (size_t)b * Cc * HWn + n;

    for (int half = 0; half < 2; ++half) {
        for (int i = tid; i < 128 * 32; i += 256) {
            int c = i >> 5, d = i & 31;
            int cg = half * 128 + c;
            sW[0][c][d] = Wq[d * Cc + cg];
            sW[1][c][d] = Wk[d * Cc + cg];
            sW[2][c][d] = Wv[d * Cc + cg];
        }
        __syncthreads();
        for (int c = 0; c < 128; ++c) {
            float xv = fbase[(size_t)(half * 128 + c) * HWn];
#pragma unroll
            for (int d4 = 0; d4 < 8; ++d4) {
                float4 wq = *(const float4*)&sW[0][c][d4 * 4];
                float4 wk = *(const float4*)&sW[1][c][d4 * 4];
                float4 wv = *(const float4*)&sW[2][c][d4 * 4];
                aq4[d4].x += xv * wq.x; aq4[d4].y += xv * wq.y;
                aq4[d4].z += xv * wq.z; aq4[d4].w += xv * wq.w;
                ak4[d4].x += xv * wk.x; ak4[d4].y += xv * wk.y;
                ak4[d4].z += xv * wk.z; ak4[d4].w += xv * wk.w;
                av4[d4].x += xv * wv.x; av4[d4].y += xv * wv.y;
                av4[d4].z += xv * wv.z; av4[d4].w += xv * wv.w;
            }
        }
        __syncthreads();
    }
    const float* aqf = (const float*)aq4;
    const float* akf = (const float*)ak4;
    const float* avf = (const float*)av4;
    float sq = 0.f, sk = 0.f;
#pragma unroll
    for (int d = 0; d < CIM; ++d) { sq += aqf[d] * aqf[d]; sk += akf[d] * akf[d]; }
    float iq = 1.0f / fmaxf(sqrtf(sq), 1e-12f);
    float ik = 1.0f / fmaxf(sqrtf(sk), 1e-12f);

    unsigned int* pq = (unsigned int*)(qn + ((size_t)b * HWn + n) * CIM);
    unsigned int* pk = (unsigned int*)(kn + ((size_t)b * HWn + n) * CIM);
    float* pv = vout + ((size_t)b * HWn + n) * CIM;
#pragma unroll
    for (int d2 = 0; d2 < 16; ++d2) {
        pq[d2] = pack2(f2bf(aqf[2 * d2] * iq), f2bf(aqf[2 * d2 + 1] * iq));
        pk[d2] = pack2(f2bf(akf[2 * d2] * ik), f2bf(akf[2 * d2 + 1] * ik));
        pv[2 * d2]     = avf[2 * d2];
        pv[2 * d2 + 1] = avf[2 * d2 + 1];
    }
}

// ---------------------------------------------------------------------------
// Kernel 3: fused cosine-similarity attention + top-9 + small MLP + gather.
// kn column chunks (128 rows x 64 B = 8 KB) are staged into LDS ONCE PER BLOCK
// via GLOBAL_LOAD_ASYNC_TO_LDS_B128, double-buffered so the DMA of chunk i+1
// overlaps the WMMA + top-k scan of chunk i. All 4 waves share the staged data
// (4x less global traffic); B-fragments come from LDS (cheap dscnt waits).
// ---------------------------------------------------------------------------
__global__ __launch_bounds__(128)
void attn_topk_kernel(const unsigned short* __restrict__ qn,
                      const unsigned short* __restrict__ kn,
                      const float* __restrict__ vbuf,
                      const float* __restrict__ adw_p, const float* __restrict__ adb_p,
                      const float* __restrict__ pm1,  const float* __restrict__ pm2,
                      unsigned short* __restrict__ ovb)
{
    __shared__ unsigned short knc[2][128][32]; // double-buffered kn chunk (2 x 8 KB)
    __shared__ float sc[4][16][136];           // per-wave 16x128 score chunk (+pad)
    __shared__ float mgv[4][16][18];
    __shared__ int   mgi[4][16][18];

    const int tid  = threadIdx.x;
    const int wid  = tid >> 5, lane = tid & 31;
    const int lr   = lane & 15, h = lane >> 4;
    const int gw   = blockIdx.x * 4 + wid;     // 800 waves total, exact
    const int b    = gw / (HWn / 16);
    const int n0   = (gw % (HWn / 16)) * 16;

    // A fragment: qn rows n0..n0+15 (16x32 bf16, WMMA lane layout)
    FragAB afrag;
    {
        const unsigned short* qrow = qn + ((size_t)b * HWn + (n0 + lr)) * CIM;
        afrag.q[0] = *(const uint4*)(qrow + h * 8);
        afrag.q[1] = *(const uint4*)(qrow + 16 + h * 8);
    }

    float tv[9]; int ti[9];
#pragma unroll
    for (int s = 0; s < 9; ++s) { tv[s] = -3.0e38f; ti[s] = 0; }

    const unsigned short* kbase = kn + (size_t)b * HWn * CIM;
    // NOTE: all 4 waves of a block map to the same batch b (400 row-blocks per
    // batch, 4 per block, 100 blocks per batch) so one staged chunk serves all.

    // prologue: kick off DMA of chunk 0 into buffer 0 (4 b128 issues per thread)
    {
        const unsigned ldsk = (unsigned)(uintptr_t)&knc[0][0][0];
        for (int i = tid; i < 512; i += 128) {
            int row = i >> 2, q4 = i & 3;
            async_load_b128(ldsk + (unsigned)(row * 64 + q4 * 16),
                            kbase + (size_t)row * CIM + q4 * 8);
        }
    }

    for (int chunk = 0; chunk < HWn / 128; ++chunk) {
        const int cbase = chunk * 128;
        const int buf   = chunk & 1;

        wait_asynccnt0();          // this wave's issued DMAs (chunk's data) done
        __syncthreads();           // ... and everyone else's

        // ---- build 8 B-fragments from LDS (ds_load_b128 pairs) ----
        FragAB bfr[8];
#pragma unroll
        for (int t = 0; t < 8; ++t) {
            const unsigned short* krow = &knc[buf][t * 16 + lr][0];
            bfr[t].q[0] = *(const uint4*)(krow + h * 8);
            bfr[t].q[1] = *(const uint4*)(krow + 16 + h * 8);
        }
        __syncthreads();           // all waves done reading knc[buf^?]: safe reissue

        // ---- overlap: kick off DMA of next chunk into the other buffer ----
        if (chunk + 1 < HWn / 128) {
            const unsigned ldsk = (unsigned)(uintptr_t)&knc[buf ^ 1][0][0];
            const unsigned short* src = kbase + (size_t)(cbase + 128) * CIM;
            for (int i = tid; i < 512; i += 128) {
                int row = i >> 2, q4 = i & 3;
                async_load_b128(ldsk + (unsigned)(row * 64 + q4 * 16),
                                src + (size_t)row * CIM + q4 * 8);
            }
        }
        if (chunk + 2 < HWn / 128)   // warm L2 two chunks ahead
            __builtin_prefetch(kbase + (size_t)(cbase + 256 + lane * 4) * CIM, 0, 1);

        // ---- 8 WMMAs back-to-back into separate accumulators ----
        FragC accs[8];
#pragma unroll
        for (int t = 0; t < 8; ++t) {
            accs[t].v = (v8f){0.f, 0.f, 0.f, 0.f, 0.f, 0.f, 0.f, 0.f};
            accs[t].v = __builtin_amdgcn_wmma_f32_16x16x32_bf16(
                            false, afrag.v, false, bfr[t].v, (short)0, accs[t].v, false, false);
        }
        // ---- spill scores to LDS for the cross-lane top-k scan ----
#pragma unroll
        for (int t = 0; t < 8; ++t)
#pragma unroll
            for (int r = 0; r < 8; ++r)
                sc[wid][r + 8 * h][t * 16 + lr] = accs[t].f[r];
        __syncthreads();

        // lane scans row lr, column half h (float4 reads -> ds_load_b128)
        const int c0 = h * 64;
        const float4* rowp = (const float4*)&sc[wid][lr][c0];
        const int ibase = cbase + c0;
#pragma unroll 4
        for (int c4 = 0; c4 < 16; ++c4) {
            float4 xv = rowp[c4];
            topk_insert(tv, ti, xv.x, ibase + c4 * 4 + 0);
            topk_insert(tv, ti, xv.y, ibase + c4 * 4 + 1);
            topk_insert(tv, ti, xv.z, ibase + c4 * 4 + 2);
            topk_insert(tv, ti, xv.w, ibase + c4 * 4 + 3);
        }
        __syncthreads();
    }

    // merge the two half-lists per row
#pragma unroll
    for (int s = 0; s < 9; ++s) { mgv[wid][lr][h * 9 + s] = tv[s]; mgi[wid][lr][h * 9 + s] = ti[s]; }
    __syncthreads();

    if (h == 0) {
#pragma unroll
        for (int e = 9; e < 18; ++e)
            topk_insert(tv, ti, mgv[wid][lr][e], mgi[wid][lr][e]);

        // ---- per-row tail: ADConv affine, 9->18->9 MLP, softmax, gather v ----
        const float adw = adw_p[0], adb = adb_p[0];
        float xa[9], pos[9];
#pragma unroll
        for (int p = 0; p < PP; ++p) {
            float x = tv[p] * adw + adb;
            pos[p] = x > 0.f ? x : 0.f;
            xa[p]  = x > 0.f ? x : -100000.0f;
        }
        float hdn[18];
#pragma unroll
        for (int q2 = 0; q2 < 18; ++q2) {
            float s = 0.f;
#pragma unroll
            for (int p = 0; p < PP; ++p) s += pos[p] * pm1[q2 * 9 + p];
            hdn[q2] = s > 0.f ? s : 0.f;
        }
        float msk[9];
#pragma unroll
        for (int p = 0; p < PP; ++p) {
            float s = 0.f;
#pragma unroll
            for (int q2 = 0; q2 < 18; ++q2) s += hdn[q2] * pm2[p * 18 + q2];
            msk[p] = 1.0f / (1.0f + expf(-s));
        }
        float m = -3.0e38f;
#pragma unroll
        for (int p = 0; p < PP; ++p) m = fmaxf(m, xa[p]);
        float ex[9], se = 0.f;
#pragma unroll
        for (int p = 0; p < PP; ++p) { ex[p] = expf(xa[p] - m); se += ex[p]; }
        float inv = 1.0f / se;

        float ov[CIM];
#pragma unroll
        for (int d = 0; d < CIM; ++d) ov[d] = 0.f;
#pragma unroll
        for (int p = 0; p < PP; ++p) {
            float wf = ex[p] * inv * msk[p];
            const float* vr = vbuf + ((size_t)b * HWn + ti[p]) * CIM;
#pragma unroll
            for (int d = 0; d < CIM; ++d) ov[d] += wf * vr[d];
        }
        unsigned int* po = (unsigned int*)(ovb + ((size_t)b * HWn + (n0 + lr)) * CIM);
#pragma unroll
        for (int d2 = 0; d2 < 16; ++d2)
            po[d2] = pack2(f2bf(ov[2 * d2]), f2bf(ov[2 * d2 + 1]));
    }
}

// ---------------------------------------------------------------------------
// Kernel 4: x = relu(bn(W_eff@ov + W2@feat + b)); r0 = relu(bn(dsn_w@x + db)).
// feat tile staged into LDS via GLOBAL_LOAD_ASYNC_TO_LDS_B128 (ASYNCcnt path),
// two 32 KB passes; f32->bf16 conversion happens while building B-fragments.
// 9 chained WMMAs per 16-channel output tile (K = 32 ov + 8*32 feat).
// ---------------------------------------------------------------------------
__global__ __launch_bounds__(128)
void final_kernel(const float* __restrict__ feat, const unsigned short* __restrict__ ovb,
                  const unsigned short* __restrict__ Weff, const unsigned short* __restrict__ W2,
                  const float* __restrict__ scaleo, const float* __restrict__ biaso,
                  const float* __restrict__ dsn_w, const float* __restrict__ dsn_b,
                  const float* __restrict__ dsn_g, const float* __restrict__ dsn_be,
                  float* __restrict__ outx, float* __restrict__ outr)
{
    __shared__ float featf[128][64];                    // 32 KB f32 staging tile
    const int tid  = threadIdx.x;
    const int wid  = tid >> 5, lane = tid & 31;
    const int lr   = lane & 15, h = lane >> 4;
    const int b    = blockIdx.x / (HWn / 64);
    const int n0   = (blockIdx.x % (HWn / 64)) * 64;

    const float* fb = feat + (size_t)b * Cc * HWn;
    const unsigned ldsbase = (unsigned)(uintptr_t)&featf[0][0];

    const int pbase = wid * 16;
    const int px    = pbase + lr;

    // B fragment for the folded ov path (bf16 rows, direct b128 loads)
    FragAB bov;
    {
        const unsigned short* orow = ovb + ((size_t)b * HWn + n0 + px) * CIM;
        bov.q[0] = *(const uint4*)(orow + h * 8);
        bov.q[1] = *(const uint4*)(orow + 16 + h * 8);
    }

    // Stage feat channels in two async passes; build 8 feat B-fragments.
    FragAB bfeat[8];
#pragma unroll
    for (int st = 0; st < 2; ++st) {
        const float* src = fb + (size_t)(st * 128) * HWn + n0;
        for (int i = tid; i < 128 * 16; i += 128) {     // 2048 float4 tiles per pass
            int c = i >> 4, q = i & 15;
            async_load_b128(ldsbase + (unsigned)((c * 64 + q * 4) * 4),
                            src + (size_t)c * HWn + q * 4);
        }
        wait_asynccnt0();
        __syncthreads();
#pragma unroll
        for (int kk2 = 0; kk2 < 4; ++kk2) {
            const int kk = st * 4 + kk2;
            const int cb = kk2 * 32;                    // stage-local channel base
#pragma unroll
            for (int e = 0; e < 8; ++e) {
                bfeat[kk].s[e]     = f2bf(featf[cb + h * 8 + e][px]);
                bfeat[kk].s[8 + e] = f2bf(featf[cb + 16 + h * 8 + e][px]);
            }
        }
        __syncthreads();                                // safe to overwrite tile
    }

    float r0part = 0.f;
    for (int o0 = 0; o0 < Cc; o0 += 16) {
        FragC acc;
        acc.v = (v8f){0.f, 0.f, 0.f, 0.f, 0.f, 0.f, 0.f, 0.f};
        {   // ov contribution through folded W_eff
            FragAB aw;
            const unsigned short* wrow = Weff + (size_t)(o0 + lr) * CIM;
            aw.q[0] = *(const uint4*)(wrow + h * 8);
            aw.q[1] = *(const uint4*)(wrow + 16 + h * 8);
            acc.v = __builtin_amdgcn_wmma_f32_16x16x32_bf16(
                        false, aw.v, false, bov.v, (short)0, acc.v, false, false);
        }
        const unsigned short* w2row = W2 + (size_t)(o0 + lr) * 256;
#pragma unroll
        for (int kk = 0; kk < 8; ++kk) {
            FragAB a2;
            a2.q[0] = *(const uint4*)(w2row + kk * 32 + h * 8);
            a2.q[1] = *(const uint4*)(w2row + kk * 32 + 16 + h * 8);
            acc.v = __builtin_amdgcn_wmma_f32_16x16x32_bf16(
                        false, a2.v, false, bfeat[kk].v, (short)0, acc.v, false, false);
        }
        // fused bias + eval-BN + ReLU epilogue; accumulate dsn head
#pragma unroll
        for (int r = 0; r < 8; ++r) {
            int o = o0 + r + 8 * h;
            float xv = acc.f[r] * scaleo[o] + biaso[o];
            xv = xv > 0.f ? xv : 0.f;
            outx[(size_t)b * Cc * HWn + (size_t)o * HWn + n0 + px] = xv;
            r0part += dsn_w[o] * xv;
        }
    }
    r0part += __shfl_xor(r0part, 16, 32);
    if (h == 0) {
        float g  = dsn_g[0] * rsqrtf(1.0f + 1e-5f);
        float rv = (r0part + dsn_b[0]) * g + dsn_be[0];
        rv = rv > 0.f ? rv : 0.f;
        outr[(size_t)b * HWn + n0 + px] = rv;
    }
}

// ---------------------------------------------------------------------------
extern "C" void kernel_launch(void* const* d_in, const int* in_sizes, int n_in,
                              void* d_out, int out_size, void* d_ws, size_t ws_size,
                              hipStream_t stream)
{
    (void)in_sizes; (void)n_in; (void)out_size; (void)ws_size;
    const float* feat   = (const float*)d_in[0];
    const float* Wq     = (const float*)d_in[1];
    const float* Wk     = (const float*)d_in[2];
    const float* Wv     = (const float*)d_in[3];
    const float* adw    = (const float*)d_in[4];
    const float* adb    = (const float*)d_in[5];
    const float* pm1    = (const float*)d_in[6];
    const float* pm2    = (const float*)d_in[7];
    const float* back_w = (const float*)d_in[8];
    const float* catt   = (const float*)d_in[9];
    const float* catt1  = (const float*)d_in[10];
    const float* mlp_w  = (const float*)d_in[11];
    const float* mlp_b  = (const float*)d_in[12];
    const float* mlp_g  = (const float*)d_in[13];
    const float* mlp_be = (const float*)d_in[14];
    const float* dsn_w  = (const float*)d_in[15];
    const float* dsn_b  = (const float*)d_in[16];
    const float* dsn_g  = (const float*)d_in[17];
    const float* dsn_be = (const float*)d_in[18];

    char* ws = (char*)d_ws;
    unsigned short* qn   = (unsigned short*)(ws + 0);          //  819,200 B
    unsigned short* kn   = (unsigned short*)(ws + 819200);     //  819,200 B
    float*          vbuf = (float*)(ws + 1638400);             // 1,638,400 B
    unsigned short* ovb  = (unsigned short*)(ws + 3276800);    //  819,200 B
    unsigned short* Weff = (unsigned short*)(ws + 4096000);    //   16,384 B
    unsigned short* W2   = (unsigned short*)(ws + 4112384);    //  131,072 B
    float*          sco  = (float*)(ws + 4243456);             //    1,024 B
    float*          bio  = (float*)(ws + 4244480);             //    1,024 B

    float* outx = (float*)d_out;
    float* outr = outx + (size_t)Bb * Cc * HWn;

    precompute_kernel<<<256, 256, 0, stream>>>(mlp_w, catt, catt1, back_w,
                                               mlp_b, mlp_g, mlp_be,
                                               Weff, W2, sco, bio);
    qkv_kernel<<<(Bb * HWn) / 256, 256, 0, stream>>>(feat, Wq, Wk, Wv, qn, kn, vbuf);
    attn_topk_kernel<<<(Bb * HWn / 16) / 4, 128, 0, stream>>>(qn, kn, vbuf,
                                                              adw, adb, pm1, pm2, ovb);
    final_kernel<<<(Bb * HWn) / 64, 128, 0, stream>>>(feat, ovb, Weff, W2, sco, bio,
                                                      dsn_w, dsn_b, dsn_g, dsn_be,
                                                      outx, outr);
}
